// GCN_7524782702754
// MI455X (gfx1250) — matrix operand order repaired and checked by reading, not compiled, not run on hardware
//
#include <hip/hip_runtime.h>

#define DF 128          // feature width
#define LN_EPS 1e-5f

typedef __attribute__((ext_vector_type(16))) __bf16 v16bf;
typedef __attribute__((ext_vector_type(8)))  __bf16 v8bf;
typedef __attribute__((ext_vector_type(8)))  float  v8f;

// ---------------- utility kernels ----------------

__global__ void gcn_zero4_kernel(float4* __restrict__ p, int n4) {
    int i = blockIdx.x * blockDim.x + threadIdx.x;
    if (i < n4) p[i] = make_float4(0.f, 0.f, 0.f, 0.f);
}

__global__ void gcn_deg_kernel(const int* __restrict__ src, const int* __restrict__ dst,
                               float* __restrict__ degOut, float* __restrict__ degIn, int nE) {
    int e = blockIdx.x * blockDim.x + threadIdx.x;
    if (e < nE) {
        unsafeAtomicAdd(&degOut[src[e]], 1.0f);
        unsafeAtomicAdd(&degIn [dst[e]], 1.0f);
    }
}

// deg -> 1/sqrt(max(deg,1)), in place over both arrays (contiguous 2N)
__global__ void gcn_rsqrt_kernel(float* __restrict__ p, int n) {
    int i = blockIdx.x * blockDim.x + threadIdx.x;
    if (i < n) p[i] = rsqrtf(fmaxf(p[i], 1.0f));
}

// ---------------- edge scatter: agg[dst] += h[src] * iso[src] ----------------
// one edge per wave32; each lane handles 4 consecutive floats (float4 load + 4 f32 atomics)
__global__ __launch_bounds__(256)
void gcn_scatter_kernel(const float* __restrict__ h, const int* __restrict__ src,
                        const int* __restrict__ dst, const float* __restrict__ iso,
                        float* __restrict__ agg, int nE) {
    const int t = blockIdx.x * blockDim.x + threadIdx.x;
    const int e = t >> 5;
    if (e >= nE) return;
    const int lane = t & 31;
    const int s = src[e];
    const int d = dst[e];
    const float sc = iso[s];
    const float4 v = *(const float4*)&h[(size_t)s * DF + lane * 4];
    float* base = &agg[(size_t)d * DF + lane * 4];
    unsafeAtomicAdd(base + 0, v.x * sc);
    unsafeAtomicAdd(base + 1, v.y * sc);
    unsafeAtomicAdd(base + 2, v.z * sc);
    unsafeAtomicAdd(base + 3, v.w * sc);
}

// ---------------- fused (isi-scale) x W + b -> LayerNorm -> ReLU ----------------
// Block = 256 threads (8 waves). Each block computes a 16-row x 128-col output tile.
// Wave w computes columns [16w, 16w+16) with v_wmma_f32_16x16x32_bf16, K-loop of 4.
__global__ __launch_bounds__(256)
void gcn_gemm_ln_kernel(const float* __restrict__ agg, const float* __restrict__ isi,
                        const float* __restrict__ W, const float* __restrict__ bias,
                        const float* __restrict__ gamma, const float* __restrict__ beta,
                        float* __restrict__ out, int nRows) {
    __shared__ __bf16 Als[16][DF];   // A tile, bf16, isi-scaled
    __shared__ float  Ots[16][DF];   // GEMM result + bias, fp32

    const int tid  = threadIdx.x;
    const int lane = tid & 31;
    const int wave = tid >> 5;       // 0..7
    const int r0   = blockIdx.x * 16;

    // ---- stage A tile: each thread loads 8 consecutive floats of one row ----
    {
        const int row = tid >> 4;            // 0..15
        const int col = (tid & 15) * 8;      // 0,8,...,120
        const int gr  = r0 + row;
        float4 p0 = make_float4(0.f, 0.f, 0.f, 0.f), p1 = p0;
        float s = 0.f;
        if (gr < nRows) {
            s  = isi[gr];
            p0 = *(const float4*)&agg[(size_t)gr * DF + col];
            p1 = *(const float4*)&agg[(size_t)gr * DF + col + 4];
        }
        __bf16* dp = &Als[row][col];
        dp[0] = (__bf16)(p0.x * s); dp[1] = (__bf16)(p0.y * s);
        dp[2] = (__bf16)(p0.z * s); dp[3] = (__bf16)(p0.w * s);
        dp[4] = (__bf16)(p1.x * s); dp[5] = (__bf16)(p1.y * s);
        dp[6] = (__bf16)(p1.z * s); dp[7] = (__bf16)(p1.w * s);
    }
    __syncthreads();

    // ---- B fragments for this wave's 16 columns (ISA B layout: lane%16=N, lane/16=K-half) ----
    const int colBase = wave * 16;
    const int nIdx = lane & 15;
    const int kh   = lane >> 4;
    v16bf bfrag[4];
#pragma unroll
    for (int kc = 0; kc < 4; ++kc) {
#pragma unroll
        for (int i = 0; i < 16; ++i) {
            const int k = kc * 32 + kh * 8 + (i < 8 ? i : i + 8);
            bfrag[kc][i] = (__bf16)W[(size_t)k * DF + colBase + nIdx];
        }
    }

    // ---- A fragments from LDS + WMMA K-loop ----
    v8f acc = {};
    const int m = nIdx;              // A row = lane%16 in both lane halves
#pragma unroll
    for (int kc = 0; kc < 4; ++kc) {
        const int kbase = kc * 32 + kh * 8;      // lane-half selects K 0-7 vs 8-15
        v8bf lo = *(const v8bf*)&Als[m][kbase];        // K = kbase..kbase+7
        v8bf hi = *(const v8bf*)&Als[m][kbase + 16];   // K = kbase+16..kbase+23
        v16bf a = __builtin_shufflevector(lo, hi, 0, 1, 2, 3, 4, 5, 6, 7,
                                                  8, 9, 10, 11, 12, 13, 14, 15);
        acc = __builtin_amdgcn_wmma_f32_16x16x32_bf16(
            /*neg_a=*/false, a, /*neg_b=*/false, bfrag[kc],
            /*c_mod=*/(short)0, acc, /*reuse_a=*/false, /*reuse_b=*/false);
    }

    // ---- scatter WMMA result (+bias) to LDS (D layout: lane%16=N, rows lane/16*8 + r) ----
    {
        const int c  = colBase + nIdx;
        const float bb = bias[c];
        const int rbase = kh * 8;
#pragma unroll
        for (int r = 0; r < 8; ++r) Ots[rbase + r][c] = acc[r] + bb;
    }
    __syncthreads();

    // ---- LayerNorm + ReLU: 16 threads per row, 8 values per thread ----
    {
        const int row = tid >> 4;        // 0..15
        const int g   = tid & 15;        // 16-lane group inside half-wave
        float v[8], s = 0.f, ss = 0.f;
#pragma unroll
        for (int i = 0; i < 8; ++i) {
            v[i] = Ots[row][g * 8 + i];
            s  += v[i];
            ss += v[i] * v[i];
        }
#pragma unroll
        for (int mask = 1; mask <= 8; mask <<= 1) {
            s  += __shfl_xor(s,  mask, 32);
            ss += __shfl_xor(ss, mask, 32);
        }
        const float mean = s * (1.0f / DF);
        const float var  = ss * (1.0f / DF) - mean * mean;
        const float inv  = rsqrtf(var + LN_EPS);
        const int gr = r0 + row;
        if (gr < nRows) {
#pragma unroll
            for (int i = 0; i < 8; ++i) {
                const int c = g * 8 + i;
                float y = (v[i] - mean) * inv * gamma[c] + beta[c];
                out[(size_t)gr * DF + c] = fmaxf(y, 0.0f);
            }
        }
    }
}

// ---------------- launcher ----------------

extern "C" void kernel_launch(void* const* d_in, const int* in_sizes, int n_in,
                              void* d_out, int out_size, void* d_ws, size_t ws_size,
                              hipStream_t stream) {
    const float* feat = (const float*)d_in[0];
    const int*   src  = (const int*)  d_in[1];
    const int*   dst  = (const int*)  d_in[2];
    const float* W1   = (const float*)d_in[3];
    const float* b1   = (const float*)d_in[4];
    const float* g1   = (const float*)d_in[5];
    const float* be1  = (const float*)d_in[6];
    const float* W2   = (const float*)d_in[7];
    const float* b2   = (const float*)d_in[8];
    const float* g2   = (const float*)d_in[9];
    const float* be2  = (const float*)d_in[10];
    float* out = (float*)d_out;

    const int N = in_sizes[0] / DF;     // 50000
    const int E = in_sizes[1];          // 800000

    float* iso = (float*)d_ws;          // [N] degree -> inv-sqrt (out)
    float* isi = iso + N;               // [N] degree -> inv-sqrt (in)
    float* agg = isi + N;               // [N*DF] scatter accumulator (16B aligned: 2N*4 % 16 == 0)

    const int threads = 256;

    // degrees -> inverse sqrt (shared by both layers)
    gcn_zero4_kernel<<<(2 * N / 4 + threads - 1) / threads, threads, 0, stream>>>(
        (float4*)iso, 2 * N / 4);
    gcn_deg_kernel<<<(E + threads - 1) / threads, threads, 0, stream>>>(src, dst, iso, isi, E);
    gcn_rsqrt_kernel<<<(2 * N + threads - 1) / threads, threads, 0, stream>>>(iso, 2 * N);

    const int scatterBlocks = (E * 32 + threads - 1) / threads;
    const int aggQuads = N * DF / 4;
    const int zeroBlocks = (aggQuads + threads - 1) / threads;
    const int gemmBlocks = (N + 15) / 16;

    // ---- layer 1: features -> d_out ----
    gcn_zero4_kernel<<<zeroBlocks, threads, 0, stream>>>((float4*)agg, aggQuads);
    gcn_scatter_kernel<<<scatterBlocks, threads, 0, stream>>>(feat, src, dst, iso, agg, E);
    gcn_gemm_ln_kernel<<<gemmBlocks, threads, 0, stream>>>(agg, isi, W1, b1, g1, be1, out, N);

    // ---- layer 2: d_out -> d_out (scatter fully reads h1 before GEMM rewrites it) ----
    gcn_zero4_kernel<<<zeroBlocks, threads, 0, stream>>>((float4*)agg, aggQuads);
    gcn_scatter_kernel<<<scatterBlocks, threads, 0, stream>>>(out, src, dst, iso, agg, E);
    gcn_gemm_ln_kernel<<<gemmBlocks, threads, 0, stream>>>(agg, isi, W2, b2, g2, be2, out, N);
}
